// YuanAttention_76321568850270
// MI455X (gfx1250) — matrix operand order repaired and checked by reading, not compiled, not run on hardware
//
#include <hip/hip_runtime.h>
#include <cstdint>
#include <cstddef>

typedef __attribute__((ext_vector_type(16))) __bf16 v16bf;
typedef __attribute__((ext_vector_type(8)))  float  v8f;

#define WMMA_BF16(a, b, c) \
  __builtin_amdgcn_wmma_f32_16x16x32_bf16(false, (a), false, (b), (short)0, (c), false, false)

union CvtU4 { uint4 u; __bf16 h[8]; };

// ---------------------------------------------------------------------------
// Generic tiled bf16 GEMM: C[M,N] = A[M,K] * B[K,N] (+ Cin) (+ bias[N])
// A,B row-major bf16. Output f32 (Cf) and/or bf16 (Cb).
// Block 256 threads = 8 waves; tile 128x128, K-step 32. Each wave: 32x64.
// A staged row-major (padded), B staged TRANSPOSED so fragment reads are
// contiguous ds_load_b128. Tiles software-pipelined through registers.
// ---------------------------------------------------------------------------
#define AS_STR 40   // 128x32 A tile, padded row stride (20 dwords: conflict-free)
#define BT_STR 40   // 128x32 B^T tile (N-major), padded

__global__ __launch_bounds__(256)
void gemm_bf16_kernel(const __bf16* __restrict__ A, const __bf16* __restrict__ B,
                      const float* bias, const float* Cin,
                      float* Cf, __bf16* Cb, int M, int N, int K) {
  __shared__ __align__(16) __bf16 As[128 * AS_STR];
  __shared__ __align__(16) __bf16 Bt[128 * BT_STR];

  const int tid  = threadIdx.x;
  const int w    = tid >> 5;
  const int lane = tid & 31;
  const int half = lane >> 4;
  const int ln   = lane & 15;
  const int bm   = blockIdx.y * 128;
  const int bn   = blockIdx.x * 128;
  const int wr   = (w >> 1) * 32;
  const int wc   = (w & 1) * 64;

  // Chunk geometry: A tile = 512 x (8 bf16), B tile = 512 x (8 bf16).
  const int arow = tid >> 2, acol = (tid & 3) * 8;       // + second chunk at arow+64
  const int brow = tid >> 4, bcol = (tid & 15) * 8;      // + second chunk at brow+16

  v8f acc[2][4];
#pragma unroll
  for (int mi = 0; mi < 2; ++mi)
#pragma unroll
    for (int ni = 0; ni < 4; ++ni)
#pragma unroll
      for (int r = 0; r < 8; ++r) acc[mi][ni][r] = 0.0f;

  const int nk = K >> 5;
  uint4 ra0, ra1, rb0, rb1;
  auto load_tile = [&](int k0) {
    ra0 = *(const uint4*)&A[(size_t)(bm + arow) * K + k0 + acol];
    ra1 = *(const uint4*)&A[(size_t)(bm + arow + 64) * K + k0 + acol];
    rb0 = *(const uint4*)&B[(size_t)(k0 + brow) * N + bn + bcol];
    rb1 = *(const uint4*)&B[(size_t)(k0 + brow + 16) * N + bn + bcol];
  };
  load_tile(0);

  for (int kt = 0; kt < nk; ++kt) {
    // Commit the pre-loaded tile to LDS (B transposed: Bt[n][k]).
    *(uint4*)&As[arow * AS_STR + acol] = ra0;
    *(uint4*)&As[(arow + 64) * AS_STR + acol] = ra1;
    {
      CvtU4 c0, c1; c0.u = rb0; c1.u = rb1;
#pragma unroll
      for (int i = 0; i < 8; ++i) {
        Bt[(bcol + i) * BT_STR + brow] = c0.h[i];
        Bt[(bcol + i) * BT_STR + brow + 16] = c1.h[i];
      }
    }
    __syncthreads();
    if (kt + 1 < nk) load_tile((kt + 1) * 32);   // overlap next tile with WMMA
    if (kt + 2 < nk) {                           // global_prefetch_b8 two ahead
      __builtin_prefetch(&A[(size_t)(bm + arow) * K + (kt + 2) * 32 + acol], 0, 1);
      __builtin_prefetch(&B[(size_t)((kt + 2) * 32 + brow) * N + bn + bcol], 0, 1);
    }

    // Fragments: contiguous 8/16-element runs -> ds_load_b128.
    v16bf af[2], bfr[4];
#pragma unroll
    for (int mi = 0; mi < 2; ++mi) {
      int row = wr + mi * 16 + ln;
      int kb = half * 16;
#pragma unroll
      for (int i = 0; i < 8; ++i) {
        af[mi][i]     = As[row * AS_STR + kb + i];
        af[mi][8 + i] = As[row * AS_STR + kb + 8 + i];
      }
    }
#pragma unroll
    for (int ni = 0; ni < 4; ++ni) {
      int col = wc + ni * 16 + ln;
      int kb = half * 16;
#pragma unroll
      for (int i = 0; i < 16; ++i) bfr[ni][i] = Bt[col * BT_STR + kb + i];
    }
#pragma unroll
    for (int mi = 0; mi < 2; ++mi)
#pragma unroll
      for (int ni = 0; ni < 4; ++ni)
        acc[mi][ni] = WMMA_BF16(af[mi], bfr[ni], acc[mi][ni]);
    __syncthreads();
  }

  // Epilogue: D layout -> row = r + 8*half, col = ln within each 16x16 tile.
#pragma unroll
  for (int mi = 0; mi < 2; ++mi)
#pragma unroll
    for (int ni = 0; ni < 4; ++ni) {
      int col = bn + wc + ni * 16 + ln;
#pragma unroll
      for (int r = 0; r < 8; ++r) {
        int row = bm + wr + mi * 16 + half * 8 + r;
        float v = acc[mi][ni][r];
        if (Cin)  v += Cin[(size_t)row * N + col];
        if (bias) v += bias[col];
        if (Cf) Cf[(size_t)row * N + col] = v;
        if (Cb) Cb[(size_t)row * N + col] = (__bf16)v;
      }
    }
}

// ---------------------------------------------------------------------------
// fp32 -> bf16 convert, optionally shifted one row back (causal conv tap 0).
// ---------------------------------------------------------------------------
__global__ void cvt_shift_kernel(const float* __restrict__ src, __bf16* __restrict__ dst,
                                 int rows, int cols, int shift, int batchlen) {
  int idx = blockIdx.x * blockDim.x + threadIdx.x;
  if (idx >= rows * cols) return;
  int row = idx / cols;
  float v;
  if (shift) v = ((row % batchlen) == 0) ? 0.0f : src[idx - cols];
  else       v = src[idx];
  dst[idx] = (__bf16)v;
}

// conv weight [OC][IC][2] -> two GEMM-B operands [IC][OC] bf16 (per tap).
__global__ void conv_tap_kernel(const float* __restrict__ w,
                                __bf16* __restrict__ t0, __bf16* __restrict__ t1,
                                int OC, int IC) {
  int idx = blockIdx.x * blockDim.x + threadIdx.x;
  if (idx >= OC * IC) return;
  int oc = idx / IC, ic = idx % IC;
  t0[ic * OC + oc] = (__bf16)w[(size_t)idx * 2 + 0];
  t1[ic * OC + oc] = (__bf16)w[(size_t)idx * 2 + 1];
}

// RoPE cos/sin tables [S][128].
__global__ void rope_tables_kernel(const int* __restrict__ positions,
                                   float* __restrict__ cosT, float* __restrict__ sinT, int S) {
  int idx = blockIdx.x * blockDim.x + threadIdx.x;
  if (idx >= S * 128) return;
  int pos = idx >> 7, i = idx & 127, j = i & 63;
  float inv = __expf(-__logf(10000.0f) * (float)(2 * j) * (1.0f / 128.0f));
  float f = (float)positions[pos] * inv;
  cosT[idx] = __cosf(f);
  sinT[idx] = __sinf(f);
}

// Residual + RMSNorm -> bf16 H. One block per token row.
__global__ __launch_bounds__(256)
void rmsnorm_kernel(const float* __restrict__ O2, const float* __restrict__ X,
                    const float* __restrict__ wgt, __bf16* __restrict__ Hbf) {
  __shared__ float red[256];
  int row = blockIdx.x, tid = threadIdx.x;
  const float* o = O2 + (size_t)row * 2048;
  const float* x = X + (size_t)row * 2048;
  float y[8], p = 0.0f;
#pragma unroll
  for (int i = 0; i < 8; ++i) {
    int c = tid + i * 256;
    y[i] = o[c] + x[c];
    p += y[i] * y[i];
  }
  red[tid] = p;
  __syncthreads();
  for (int st = 128; st > 0; st >>= 1) {
    if (tid < st) red[tid] += red[tid + st];
    __syncthreads();
  }
  float inv = rsqrtf(red[0] * (1.0f / 2048.0f) + 1e-6f);
#pragma unroll
  for (int i = 0; i < 8; ++i) {
    int c = tid + i * 256;
    Hbf[(size_t)row * 2048 + c] = (__bf16)(y[i] * inv * wgt[c]);
  }
}

// Yuan concat-quirk + RoPE + pack into head-major [B*H][S][128] bf16.
__global__ void rope_pack_kernel(const float* __restrict__ Qf, const float* __restrict__ Kf,
                                 const __bf16* __restrict__ Vbf,
                                 const float* __restrict__ cosT, const float* __restrict__ sinT,
                                 const int* __restrict__ positions,
                                 __bf16* __restrict__ Qh, __bf16* __restrict__ Kh,
                                 __bf16* __restrict__ Vh, int S) {
  int idx = blockIdx.x * blockDim.x + threadIdx.x;
  if (idx >= 32 * S * 128) return;
  int d  = idx & 127;
  int t  = (idx >> 7) % S;
  int bh = idx / (S * 128);
  int b = bh >> 4, h = bh & 15;
  size_t tokrow = (size_t)(b * S + t) * 2048;
  // heads 0-7 read both halves from the Q projection, 8-15 from the K projection
  const float* src = (h < 8) ? Qf : Kf;
  int hh = (h < 8) ? h : h - 8;
  int qc = hh * 256, kc = hh * 256 + 128;
  int pos = positions[t];
  float c = cosT[pos * 128 + d], s = sinT[pos * 128 + d];
  int partner = (d < 64) ? d + 64 : d - 64;
  float sgn = (d < 64) ? -1.0f : 1.0f;
  float qv = src[tokrow + qc + d] * c + sgn * src[tokrow + qc + partner] * s;
  float kv = src[tokrow + kc + d] * c + sgn * src[tokrow + kc + partner] * s;
  size_t oidx = (size_t)bh * S * 128 + (size_t)t * 128 + d;
  Qh[oidx] = (__bf16)qv;
  Kh[oidx] = (__bf16)kv;
  Vh[oidx] = Vbf[tokrow + h * 128 + d];
}

// ---------------------------------------------------------------------------
// Flash attention (causal). Grid (S/128 q-tiles, B*H). 8 waves x 16 q rows.
// Key blocks of 64. K staged row-major (padded, conflict-free b128 reads),
// V staged transposed. K/V tiles software-pipelined through registers.
// ---------------------------------------------------------------------------
#define KS_STR 136  // 64x128 K tile, padded (68 dwords/row -> banks tile cleanly)
#define VT_STR 72   // 128x64 V^T tile, padded
#define PS_STR 72   // per-wave 16x64 P tile, padded

__global__ __launch_bounds__(256)
void attn_kernel(const __bf16* __restrict__ Qh, const __bf16* __restrict__ Kh,
                 const __bf16* __restrict__ Vh, __bf16* __restrict__ Obf, int S) {
  __shared__ __align__(16) __bf16 Ks[64 * KS_STR];
  __shared__ __align__(16) __bf16 Vt[128 * VT_STR];
  __shared__ __align__(16) __bf16 Ps[8 * 16 * PS_STR];

  const int bh = blockIdx.y, qt = blockIdx.x;
  const int tid = threadIdx.x, w = tid >> 5, lane = tid & 31;
  const int half = lane >> 4, ln = lane & 15;
  const __bf16* Qb = Qh + (size_t)bh * S * 128;
  const __bf16* Kb = Kh + (size_t)bh * S * 128;
  const __bf16* Vb = Vh + (size_t)bh * S * 128;

  // This wave's Q fragments (16 rows x 128 dims = 4 k-steps), loaded once.
  v16bf qa[4];
  const int qrow = qt * 128 + w * 16 + ln;
#pragma unroll
  for (int kk = 0; kk < 4; ++kk) {
    int kb = kk * 32 + half * 16;
    const __bf16* p = Qb + (size_t)qrow * 128 + kb;
#pragma unroll
    for (int i = 0; i < 8; ++i) { qa[kk][i] = p[i]; qa[kk][8 + i] = p[8 + i]; }
  }

  float m[8], l[8];
  v8f o[8];
#pragma unroll
  for (int r = 0; r < 8; ++r) { m[r] = -1e30f; l[r] = 0.0f; }
#pragma unroll
  for (int ni = 0; ni < 8; ++ni)
#pragma unroll
    for (int r = 0; r < 8; ++r) o[ni][r] = 0.0f;

  const float scale = 0.08838834764831845f;  // 1/sqrt(128)
  const int jmax = qt * 2 + 1;
  __bf16* Pw = &Ps[w * 16 * PS_STR];

  // 64x128 tile = 1024 chunks of 8 bf16; 4 per thread for each of K and V.
  uint4 rk[4], rv[4];
  auto load_kv = [&](int j) {
#pragma unroll
    for (int cc = 0; cc < 4; ++cc) {
      int c = tid + cc * 256;
      int row = c >> 4, col8 = (c & 15) * 8;
      rk[cc] = *(const uint4*)&Kb[(size_t)(j * 64 + row) * 128 + col8];
      rv[cc] = *(const uint4*)&Vb[(size_t)(j * 64 + row) * 128 + col8];
    }
  };
  load_kv(0);

  for (int j = 0; j <= jmax; ++j) {
    // Commit pre-loaded K (row-major) and V (transposed) tiles to LDS.
#pragma unroll
    for (int cc = 0; cc < 4; ++cc) {
      int c = tid + cc * 256;
      int row = c >> 4, col8 = (c & 15) * 8;
      *(uint4*)&Ks[row * KS_STR + col8] = rk[cc];
      CvtU4 cv; cv.u = rv[cc];
#pragma unroll
      for (int i = 0; i < 8; ++i) Vt[(col8 + i) * VT_STR + row] = cv.h[i];
    }
    __syncthreads();
    if (j < jmax) load_kv(j + 1);   // overlap next K/V tile with the math

    // S = Q * K^T over 4 key tiles of 16 (contiguous Ks reads).
    v8f s[4];
#pragma unroll
    for (int ni = 0; ni < 4; ++ni)
#pragma unroll
      for (int r = 0; r < 8; ++r) s[ni][r] = 0.0f;
#pragma unroll
    for (int kk = 0; kk < 4; ++kk) {
#pragma unroll
      for (int ni = 0; ni < 4; ++ni) {
        v16bf bk;
        int key = ni * 16 + ln;
        int kb = kk * 32 + half * 16;
#pragma unroll
        for (int i = 0; i < 16; ++i) bk[i] = Ks[key * KS_STR + kb + i];
        s[ni] = WMMA_BF16(qa[kk], bk, s[ni]);
      }
    }

    // Scale + causal mask + online softmax (rows split across half-waves).
    float rowm[8];
#pragma unroll
    for (int r = 0; r < 8; ++r) rowm[r] = -1e30f;
    const int qg0 = qt * 128 + w * 16 + half * 8;
#pragma unroll
    for (int ni = 0; ni < 4; ++ni) {
      int kg = j * 64 + ni * 16 + ln;
#pragma unroll
      for (int r = 0; r < 8; ++r) {
        float v = s[ni][r] * scale;
        if (kg > qg0 + r) v = -1e30f;
        s[ni][r] = v;
        rowm[r] = fmaxf(rowm[r], v);
      }
    }
#pragma unroll
    for (int off = 1; off < 16; off <<= 1)
#pragma unroll
      for (int r = 0; r < 8; ++r)
        rowm[r] = fmaxf(rowm[r], __shfl_xor(rowm[r], off, 16));

    float alpha[8];
#pragma unroll
    for (int r = 0; r < 8; ++r) {
      float mn = fmaxf(m[r], rowm[r]);
      alpha[r] = __expf(m[r] - mn);
      m[r] = mn;
      l[r] *= alpha[r];
    }
#pragma unroll
    for (int ni = 0; ni < 8; ++ni)
#pragma unroll
      for (int r = 0; r < 8; ++r) o[ni][r] *= alpha[r];

    float rs[8];
#pragma unroll
    for (int r = 0; r < 8; ++r) rs[r] = 0.0f;
#pragma unroll
    for (int ni = 0; ni < 4; ++ni)
#pragma unroll
      for (int r = 0; r < 8; ++r) {
        float p = __expf(s[ni][r] - m[r]);
        s[ni][r] = p;
        rs[r] += p;
      }
#pragma unroll
    for (int off = 1; off < 16; off <<= 1)
#pragma unroll
      for (int r = 0; r < 8; ++r) rs[r] += __shfl_xor(rs[r], off, 16);
#pragma unroll
    for (int r = 0; r < 8; ++r) l[r] += rs[r];

    // P (D layout) -> per-wave LDS -> A-fragment layout (wave-local, in order).
#pragma unroll
    for (int ni = 0; ni < 4; ++ni)
#pragma unroll
      for (int r = 0; r < 8; ++r)
        Pw[(r + half * 8) * PS_STR + ni * 16 + ln] = (__bf16)s[ni][r];

    // O += P * V over the 8 head-dim tiles (contiguous Vt reads).
#pragma unroll
    for (int kk = 0; kk < 2; ++kk) {
      v16bf pa;
      int kb = kk * 32 + half * 16;
#pragma unroll
      for (int i = 0; i < 8; ++i) {
        pa[i]     = Pw[ln * PS_STR + kb + i];
        pa[8 + i] = Pw[ln * PS_STR + kb + 8 + i];
      }
#pragma unroll
      for (int ni = 0; ni < 8; ++ni) {
        v16bf vb;
        int dim = ni * 16 + ln;
        int kb2 = kk * 32 + half * 16;
#pragma unroll
        for (int i = 0; i < 16; ++i) vb[i] = Vt[dim * VT_STR + kb2 + i];
        o[ni] = WMMA_BF16(pa, vb, o[ni]);
      }
    }
    __syncthreads();
  }

  // Epilogue: normalize and scatter to [B*S][H*128] bf16 for the out-proj GEMM.
  const int b = bh >> 4, h = bh & 15;
#pragma unroll
  for (int ni = 0; ni < 8; ++ni)
#pragma unroll
    for (int r = 0; r < 8; ++r) {
      int row = qt * 128 + w * 16 + half * 8 + r;
      float val = o[ni][r] / l[r];
      Obf[(size_t)(b * S + row) * 2048 + h * 128 + ni * 16 + ln] = (__bf16)val;
    }
}

// ---------------------------------------------------------------------------
static inline void launch_gemm(const __bf16* A, const __bf16* B, const float* bias,
                               const float* Cin, float* Cf, __bf16* Cb,
                               int M, int N, int K, hipStream_t s) {
  dim3 g(N / 128, M / 128);
  gemm_bf16_kernel<<<g, 256, 0, s>>>(A, B, bias, Cin, Cf, Cb, M, N, K);
}

extern "C" void kernel_launch(void* const* d_in, const int* in_sizes, int n_in,
                              void* d_out, int out_size, void* d_ws, size_t ws_size,
                              hipStream_t stream) {
  (void)in_sizes; (void)n_in; (void)out_size; (void)ws_size;
  const float* hidden = (const float*)d_in[0];
  const int*   positions = (const int*)d_in[1];
  const float* wq  = (const float*)d_in[2];
  const float* wk  = (const float*)d_in[3];
  const float* wv  = (const float*)d_in[4];
  const float* wo  = (const float*)d_in[5];
  const float* c1w = (const float*)d_in[6];
  const float* c1b = (const float*)d_in[7];
  const float* c2w = (const float*)d_in[8];
  const float* c2b = (const float*)d_in[9];
  const float* rmsw = (const float*)d_in[10];
  float* out = (float*)d_out;

  const int B = 2, S = 2048, T = B * S, HID = 2048, HALF = 1024;

  char* ws = (char*)d_ws;
  size_t off = 0;
  auto alloc = [&](size_t bytes) -> void* {
    void* p = ws + off;
    off += (bytes + 255) & ~(size_t)255;
    return p;
  };
  float*  cosT  = (float*)alloc((size_t)S * 128 * 4);
  float*  sinT  = (float*)alloc((size_t)S * 128 * 4);
  __bf16* Xbf   = (__bf16*)alloc((size_t)T * HID * 2);   // later reused as Qf (with Xprev)
  __bf16* Xprev = (__bf16*)alloc((size_t)T * HID * 2);
  __bf16* Wqb   = (__bf16*)alloc((size_t)HID * HID * 2);
  __bf16* Wkb   = (__bf16*)alloc((size_t)HID * HID * 2);
  __bf16* Wvb   = (__bf16*)alloc((size_t)HID * HID * 2);
  __bf16* Wob   = (__bf16*)alloc((size_t)HID * HID * 2);
  __bf16* B1t0  = (__bf16*)alloc((size_t)HID * HALF * 2);
  __bf16* B1t1  = (__bf16*)alloc((size_t)HID * HALF * 2);
  __bf16* B2t0  = (__bf16*)alloc((size_t)HALF * HID * 2);
  __bf16* B2t1  = (__bf16*)alloc((size_t)HALF * HID * 2);
  float*  O1f   = (float*)alloc((size_t)T * HALF * 4);
  __bf16* O1bf  = (__bf16*)alloc((size_t)T * HALF * 2);
  __bf16* O1pr  = (__bf16*)alloc((size_t)T * HALF * 2);
  float*  O2f   = (float*)alloc((size_t)T * HID * 4);    // later reused as Kf
  __bf16* Hbf   = (__bf16*)alloc((size_t)T * HID * 2);   // later reused as attention output
  __bf16* Vbf   = (__bf16*)alloc((size_t)T * HID * 2);
  __bf16* Qh    = (__bf16*)alloc((size_t)32 * S * 128 * 2);
  __bf16* Kh    = (__bf16*)alloc((size_t)32 * S * 128 * 2);
  __bf16* Vh    = (__bf16*)alloc((size_t)32 * S * 128 * 2);
  float*  Qf    = (float*)Xbf;   // 33.5MB region (Xbf+Xprev), free by then
  float*  Kf    = O2f;           // free after rmsnorm
  __bf16* Abf   = Hbf;           // free after Q/K GEMMs

  const int TB = 256;
  auto grid1 = [](int n) { return dim3((unsigned)((n + 255) / 256)); };

  // RoPE tables + bf16 conversions.
  rope_tables_kernel<<<grid1(S * 128), TB, 0, stream>>>(positions, cosT, sinT, S);
  cvt_shift_kernel<<<grid1(T * HID), TB, 0, stream>>>(hidden, Xbf, T, HID, 0, S);
  cvt_shift_kernel<<<grid1(T * HID), TB, 0, stream>>>(hidden, Xprev, T, HID, 1, S);
  cvt_shift_kernel<<<grid1(HID * HID), TB, 0, stream>>>(wq, Wqb, HID, HID, 0, S);
  cvt_shift_kernel<<<grid1(HID * HID), TB, 0, stream>>>(wk, Wkb, HID, HID, 0, S);
  cvt_shift_kernel<<<grid1(HID * HID), TB, 0, stream>>>(wv, Wvb, HID, HID, 0, S);
  cvt_shift_kernel<<<grid1(HID * HID), TB, 0, stream>>>(wo, Wob, HID, HID, 0, S);
  conv_tap_kernel<<<grid1(HALF * HID), TB, 0, stream>>>(c1w, B1t0, B1t1, HALF, HID);
  conv_tap_kernel<<<grid1(HID * HALF), TB, 0, stream>>>(c2w, B2t0, B2t1, HID, HALF);

  // V projection from raw hidden states (pre-gate), stored bf16.
  launch_gemm(Xbf, Wvb, nullptr, nullptr, nullptr, Vbf, T, HID, HID, stream);

  // LF gate: conv1 (two taps as accumulated GEMMs) -> conv2 -> residual+RMSNorm.
  launch_gemm(Xprev, B1t0, nullptr, nullptr, O1f, nullptr, T, HALF, HID, stream);
  launch_gemm(Xbf,   B1t1, c1b,     O1f,    O1f, nullptr, T, HALF, HID, stream);
  cvt_shift_kernel<<<grid1(T * HALF), TB, 0, stream>>>(O1f, O1bf, T, HALF, 0, S);
  cvt_shift_kernel<<<grid1(T * HALF), TB, 0, stream>>>(O1f, O1pr, T, HALF, 1, S);
  launch_gemm(O1pr, B2t0, nullptr, nullptr, O2f, nullptr, T, HID, HALF, stream);
  launch_gemm(O1bf, B2t1, c2b,     O2f,    O2f, nullptr, T, HID, HALF, stream);
  rmsnorm_kernel<<<dim3((unsigned)T), TB, 0, stream>>>(O2f, hidden, rmsw, Hbf);

  // Q/K projections (f32 out, quirk+RoPE applied next), then head packing.
  launch_gemm(Hbf, Wqb, nullptr, nullptr, Qf, nullptr, T, HID, HID, stream);
  launch_gemm(Hbf, Wkb, nullptr, nullptr, Kf, nullptr, T, HID, HID, stream);
  rope_pack_kernel<<<grid1(32 * S * 128), TB, 0, stream>>>(Qf, Kf, Vbf, cosT, sinT,
                                                           positions, Qh, Kh, Vh, S);

  // Causal flash attention, then output projection to f32 d_out.
  attn_kernel<<<dim3(S / 128, 32), TB, 0, stream>>>(Qh, Kh, Vh, Abf, S);
  launch_gemm(Abf, Wob, nullptr, nullptr, out, nullptr, T, HID, HID, stream);
}
